// SelfAttention1_43516608643277
// MI455X (gfx1250) — matrix operand order repaired and checked by reading, not compiled, not run on hardware
//
#include <hip/hip_runtime.h>

typedef __attribute__((ext_vector_type(16))) __bf16        v16bf;
typedef __attribute__((ext_vector_type(8)))  float         v8f;
typedef __attribute__((ext_vector_type(4)))  unsigned int  u32x4;
typedef __attribute__((ext_vector_type(2)))  unsigned int  u32x2;
typedef __attribute__((ext_vector_type(4)))  float         f32x4;

namespace {
constexpr int BATCH = 8;
constexpr int SEQ   = 2048;
constexpr int DIM   = 512;
constexpr float SCALE = 0.044194173824159216f;  // 1/sqrt(512)

// ---- attention kernel dynamic-LDS layout (bytes) ----
constexpr int QS_LD = 520;   // ushorts/row: 512 + 8 pad (stride 1040B, bank-safe)
constexpr int SB_LD = 132;   // floats/row of scores
constexpr int PB_LD = 136;   // ushorts/row of probabilities (272B stride)
constexpr int KB_LD = 72;    // ushorts/row of K chunk (144B stride)
constexpr int VT_LD = 40;    // ushorts/row of V-transposed (80B stride)
constexpr unsigned QS_OFF = 0;
constexpr unsigned SB_OFF = QS_OFF + 16 * QS_LD * 2;   // 16640
constexpr unsigned PB_OFF = SB_OFF + 16 * SB_LD * 4;   // 25088
constexpr unsigned ST_OFF = PB_OFF + 16 * PB_LD * 2;   // 29440
constexpr unsigned KV_OFF = ST_OFF + 3 * 16 * 4;       // 29632 (Kb 18432B / Vt 40960B union)
constexpr unsigned SMEM_BYTES = KV_OFF + 512 * VT_LD * 2;  // 70592
}  // namespace

union FragAB { v16bf bf; u32x4 q[2]; };

__device__ __forceinline__ unsigned short f2bf(float f) {
  union { float f; unsigned int u; } cv; cv.f = f;
  unsigned int u = cv.u;
  return (unsigned short)((u + 0x7fffu + ((u >> 16) & 1u)) >> 16);  // RNE
}

__device__ __forceinline__ unsigned int pack2bf(float lo, float hi) {
  return (unsigned int)f2bf(lo) | ((unsigned int)f2bf(hi) << 16);
}

__device__ __forceinline__ v8f zero8() {
  v8f z = {0.f, 0.f, 0.f, 0.f, 0.f, 0.f, 0.f, 0.f};
  return z;
}

// Per-lane async copy of 8 bytes global -> LDS (CDNA5, tracked on ASYNCcnt).
__device__ __forceinline__ void async_copy_b64(void* lds_dst, const void* gsrc) {
  unsigned int       lds  = (unsigned int)(size_t)lds_dst;      // LDS byte offset in addr[31:0]
  unsigned long long gptr = (unsigned long long)(size_t)gsrc;
  asm volatile("global_load_async_to_lds_b64 %0, %1, off"
               :: "v"(lds), "v"(gptr) : "memory");
}

__device__ __forceinline__ void async_wait0() {
  asm volatile("s_wait_asynccnt 0x0" ::: "memory");
}

// ---------------------------------------------------------------------------
// Kernel 1: Out[bf16] = X[fp32] @ W[fp32] + bias.  M=16384, K=N=512.
// Block = 256 threads (8 waves), tile 128x128, BK=32.
// Wave (wm,wn) = (w>>1, w&1) -> 32x64 sub-tile = 2x4 WMMA accumulators.
// ---------------------------------------------------------------------------
__global__ __launch_bounds__(256)
void proj_gemm_bf16_kernel(const float* __restrict__ X,
                           const float* __restrict__ W,
                           const float* __restrict__ bias,
                           unsigned short* __restrict__ Out) {
  constexpr int LDA = 40;  // ushorts per LDS row (32 + 8 pad -> 80B stride)
  __shared__ unsigned short As[128 * LDA];  // As[m][k]
  __shared__ unsigned short Bs[128 * LDA];  // Bs[n][k] (W transposed)

  const int tid  = threadIdx.x;
  const int wave = tid >> 5;
  const int lane = tid & 31;
  const int lr   = lane & 15;
  const int hf   = lane >> 4;
  const int wm   = wave >> 1;
  const int wn   = wave & 1;
  const int rowBase = blockIdx.y * 128;
  const int colBase = blockIdx.x * 128;

  v8f acc[2][4];
#pragma unroll
  for (int mi = 0; mi < 2; ++mi)
#pragma unroll
    for (int ni = 0; ni < 4; ++ni) acc[mi][ni] = zero8();

  for (int k0 = 0; k0 < 512; k0 += 32) {
    // stage A tile: 128x32 fp32 -> bf16 LDS (packed pair stores)
#pragma unroll
    for (int i = 0; i < 4; ++i) {
      int idx = tid + i * 256;           // float4 index, 0..1023
      int r   = idx >> 3;
      int c4  = idx & 7;
      f32x4 v = *(const f32x4*)(X + (size_t)(rowBase + r) * 512 + k0 + c4 * 4);
      unsigned int* d = (unsigned int*)&As[r * LDA + c4 * 4];
      d[0] = pack2bf(v[0], v[1]);
      d[1] = pack2bf(v[2], v[3]);
    }
    // stage B tile transposed: W[k][n] -> Bs[n][k]; process k in pairs so each
    // LDS store carries two packed bf16 along k (the contiguous LDS dim).
#pragma unroll
    for (int i = 0; i < 2; ++i) {
      int idx = tid + i * 256;           // 0..511
      int kp  = idx >> 5;                // k-pair 0..15
      int c4  = idx & 31;
      const float* w0 = W + (size_t)(k0 + kp * 2) * 512 + colBase + c4 * 4;
      f32x4 va = *(const f32x4*)(w0);
      f32x4 vb = *(const f32x4*)(w0 + 512);
      int c = c4 * 4;
      *(unsigned int*)&Bs[(c + 0) * LDA + kp * 2] = pack2bf(va[0], vb[0]);
      *(unsigned int*)&Bs[(c + 1) * LDA + kp * 2] = pack2bf(va[1], vb[1]);
      *(unsigned int*)&Bs[(c + 2) * LDA + kp * 2] = pack2bf(va[2], vb[2]);
      *(unsigned int*)&Bs[(c + 3) * LDA + kp * 2] = pack2bf(va[3], vb[3]);
    }
    __syncthreads();

    // per-lane WMMA fragments straight from ISA 16-bit A/B layout tables
    FragAB a[2], b[4];
#pragma unroll
    for (int mi = 0; mi < 2; ++mi) {
      const unsigned short* p = &As[(wm * 32 + mi * 16 + lr) * LDA + hf * 8];
      a[mi].q[0] = *(const u32x4*)(p);        // K = hf*8 .. +7
      a[mi].q[1] = *(const u32x4*)(p + 16);   // K = 16 + hf*8 .. +7
    }
#pragma unroll
    for (int ni = 0; ni < 4; ++ni) {
      const unsigned short* p = &Bs[(wn * 64 + ni * 16 + lr) * LDA + hf * 8];
      b[ni].q[0] = *(const u32x4*)(p);        // K = hf*8 .. +7
      b[ni].q[1] = *(const u32x4*)(p + 16);   // K = 16 + hf*8 .. +7
    }
#pragma unroll
    for (int mi = 0; mi < 2; ++mi)
#pragma unroll
      for (int ni = 0; ni < 4; ++ni)
        acc[mi][ni] = __builtin_amdgcn_wmma_f32_16x16x32_bf16(
            false, a[mi].bf, false, b[ni].bf, (short)0, acc[mi][ni], false, false);
    __syncthreads();
  }

  // epilogue: + bias, store bf16
#pragma unroll
  for (int ni = 0; ni < 4; ++ni) {
    int col = colBase + wn * 64 + ni * 16 + lr;
    float bv = bias[col];
#pragma unroll
    for (int mi = 0; mi < 2; ++mi) {
      int r0 = rowBase + wm * 32 + mi * 16 + hf * 8;
#pragma unroll
      for (int v = 0; v < 8; ++v)
        Out[(size_t)(r0 + v) * 512 + col] = f2bf(acc[mi][ni][v] + bv);
    }
  }
}

// ---------------------------------------------------------------------------
// Kernel 2: flash attention over bf16 projections.
// Grid (SEQ/16, BATCH), block 256 (8 waves). Each block: 16 query rows.
// Key blocks of 128: wave w owns score tile [16q x 16k(w)] and output
// feature slice [16q x 64f(w)]. Online softmax by threads 0..15.
// Q and K staging use global_load_async_to_lds_b64 (pure bf16 copies).
// ---------------------------------------------------------------------------
__global__ __launch_bounds__(256)
void attn_fused_kernel(const unsigned short* __restrict__ Qp,
                       const unsigned short* __restrict__ Kp,
                       const unsigned short* __restrict__ Vp,
                       const float* __restrict__ Mask,
                       float* __restrict__ Out) {
  extern __shared__ char smem[];
  unsigned short* Qs = (unsigned short*)(smem + QS_OFF);  // [16][520]
  float*          Sb = (float*)(smem + SB_OFF);           // [16][132]
  unsigned short* Pb = (unsigned short*)(smem + PB_OFF);  // [16][136]
  float*        mrow = (float*)(smem + ST_OFF);
  float*        lrow = mrow + 16;
  float*        crow = lrow + 16;
  unsigned short* Kb = (unsigned short*)(smem + KV_OFF);  // [128][72]  (QK^T phase)
  unsigned short* Vt = (unsigned short*)(smem + KV_OFF);  // [512][40]  (PV phase, aliased)

  const int tid  = threadIdx.x;
  const int wave = tid >> 5;
  const int lane = tid & 31;
  const int lr   = lane & 15;
  const int hf   = lane >> 4;
  const int qbase = blockIdx.x * 16;
  const int bidx  = blockIdx.y;
  const size_t batchRow = (size_t)bidx * SEQ;

  // stage the 16x512 bf16 Q block once (async global->LDS)
#pragma unroll
  for (int i = 0; i < 8; ++i) {
    int idx = tid + i * 256;  // 8-byte chunks, 2048 total
    int r   = idx >> 7;
    int c4  = idx & 127;
    async_copy_b64(&Qs[r * QS_LD + c4 * 4],
                   Qp + (batchRow + qbase + r) * DIM + c4 * 4);
  }
  if (tid < 16) { mrow[tid] = -3.0e38f; lrow[tid] = 0.0f; crow[tid] = 1.0f; }

  v8f oacc[4];
#pragma unroll
  for (int ni = 0; ni < 4; ++ni) oacc[ni] = zero8();
  async_wait0();
  __syncthreads();

  for (int kb = 0; kb < SEQ; kb += 128) {
    // ---- phase 1: S-tile = Q @ K^T (accumulate over 512 features) ----
    v8f sacc = zero8();
    for (int fc = 0; fc < DIM; fc += 64) {
      // stage K chunk: 128 keys x 64 features, async global->LDS
#pragma unroll
      for (int i = 0; i < 8; ++i) {
        int idx = tid + i * 256;  // 8-byte chunks, 2048 total
        int key = idx >> 4;
        int c4  = idx & 15;
        async_copy_b64(&Kb[key * KB_LD + c4 * 4],
                       Kp + (batchRow + kb + key) * DIM + fc + c4 * 4);
      }
      async_wait0();
      __syncthreads();
#pragma unroll
      for (int ks = 0; ks < 2; ++ks) {
        FragAB a, b;
        const unsigned short* pa = &Qs[lr * QS_LD + fc + ks * 32 + hf * 8];
        a.q[0] = *(const u32x4*)(pa);
        a.q[1] = *(const u32x4*)(pa + 16);
        const unsigned short* pb = &Kb[(wave * 16 + lr) * KB_LD + ks * 32 + hf * 8];
        b.q[0] = *(const u32x4*)(pb);
        b.q[1] = *(const u32x4*)(pb + 16);
        sacc = __builtin_amdgcn_wmma_f32_16x16x32_bf16(
            false, a.bf, false, b.bf, (short)0, sacc, false, false);
      }
      __syncthreads();
    }

    // ---- phase 2: scale + mask -> Sb ----
    {
      int kcol = wave * 16 + lr;
#pragma unroll
      for (int v = 0; v < 8; ++v) {
        int qr = v + hf * 8;
        float mval = Mask[((size_t)bidx * SEQ + qbase + qr) * SEQ + kb + kcol];
        Sb[qr * SB_LD + kcol] = sacc[v] * SCALE + mval;
      }
    }
    __syncthreads();

    // ---- phase 3: online softmax update (one thread per query row) ----
    if (tid < 16) {
      int r = tid;
      float mOld = mrow[r];
      float mBlk = -3.0e38f;
      for (int j = 0; j < 128; ++j) mBlk = fmaxf(mBlk, Sb[r * SB_LD + j]);
      float mNew = fmaxf(mOld, mBlk);
      float corr = __expf(mOld - mNew);
      float lsum = 0.0f;
      for (int j = 0; j < 128; ++j) {
        float p = __expf(Sb[r * SB_LD + j] - mNew);
        lsum += p;
        Pb[r * PB_LD + j] = f2bf(p);
      }
      mrow[r] = mNew;
      lrow[r] = lrow[r] * corr + lsum;
      crow[r] = corr;
    }
    __syncthreads();

    // ---- phase 4: rescale running output accumulators ----
#pragma unroll
    for (int v = 0; v < 8; ++v) {
      float cv = crow[v + hf * 8];
#pragma unroll
      for (int ni = 0; ni < 4; ++ni) oacc[ni][v] *= cv;
    }
    __syncthreads();

    // ---- phase 5: O += P @ V, 32 keys at a time (Vt aliases Kb region) ----
    for (int kk = 0; kk < 4; ++kk) {
      // stage V transposed: Vt[feature][key]; process keys in pairs so each
      // LDS store carries two packed bf16 along key (contiguous LDS dim).
#pragma unroll
      for (int i = 0; i < 8; ++i) {
        int idx = tid + i * 256;  // 2048 items: (key-pair, 4 features)
        int kp  = idx >> 7;       // key pair 0..15
        int c4  = idx & 127;      // feature/4
        const unsigned short* g =
            Vp + (batchRow + kb + kk * 32 + kp * 2) * DIM + c4 * 4;
        u32x2 v0 = *(const u32x2*)(g);
        u32x2 v1 = *(const u32x2*)(g + DIM);
        const unsigned short* p0 = (const unsigned short*)&v0;
        const unsigned short* p1 = (const unsigned short*)&v1;
        int f = c4 * 4;
#pragma unroll
        for (int j = 0; j < 4; ++j) {
          *(unsigned int*)&Vt[(f + j) * VT_LD + kp * 2] =
              (unsigned int)p0[j] | ((unsigned int)p1[j] << 16);
        }
      }
      __syncthreads();
      FragAB a;
      const unsigned short* pa = &Pb[lr * PB_LD + kk * 32 + hf * 8];
      a.q[0] = *(const u32x4*)(pa);
      a.q[1] = *(const u32x4*)(pa + 16);
#pragma unroll
      for (int ni = 0; ni < 4; ++ni) {
        FragAB b;
        const unsigned short* pb = &Vt[(wave * 64 + ni * 16 + lr) * VT_LD + hf * 8];
        b.q[0] = *(const u32x4*)(pb);
        b.q[1] = *(const u32x4*)(pb + 16);
        oacc[ni] = __builtin_amdgcn_wmma_f32_16x16x32_bf16(
            false, a.bf, false, b.bf, (short)0, oacc[ni], false, false);
      }
      __syncthreads();
    }
  }

  // ---- epilogue: normalize by softmax denominator, store fp32 ----
#pragma unroll
  for (int v = 0; v < 8; ++v) {
    int qr = v + hf * 8;
    float inv = 1.0f / lrow[qr];
#pragma unroll
    for (int ni = 0; ni < 4; ++ni) {
      int col = wave * 64 + ni * 16 + lr;
      Out[((size_t)bidx * SEQ + qbase + qr) * DIM + col] = oacc[ni][v] * inv;
    }
  }
}

// ---------------------------------------------------------------------------
extern "C" void kernel_launch(void* const* d_in, const int* in_sizes, int n_in,
                              void* d_out, int out_size, void* d_ws, size_t ws_size,
                              hipStream_t stream) {
  (void)in_sizes; (void)n_in; (void)out_size; (void)ws_size;
  const float* Q    = (const float*)d_in[0];
  const float* K    = (const float*)d_in[1];
  const float* V    = (const float*)d_in[2];
  const float* mask = (const float*)d_in[3];
  const float* Wq   = (const float*)d_in[4];
  const float* bq   = (const float*)d_in[5];
  const float* Wk   = (const float*)d_in[6];
  const float* bk   = (const float*)d_in[7];
  const float* Wv   = (const float*)d_in[8];
  const float* bv   = (const float*)d_in[9];
  float* out = (float*)d_out;

  // workspace: bf16 projected Q/K/V, each BATCH*SEQ*DIM elements
  unsigned short* Qp = (unsigned short*)d_ws;
  unsigned short* Kp = Qp + (size_t)BATCH * SEQ * DIM;
  unsigned short* Vp = Kp + (size_t)BATCH * SEQ * DIM;

  dim3 gemmGrid(DIM / 128, (BATCH * SEQ) / 128);
  proj_gemm_bf16_kernel<<<gemmGrid, 256, 0, stream>>>(Q, Wq, bq, Qp);
  proj_gemm_bf16_kernel<<<gemmGrid, 256, 0, stream>>>(K, Wk, bk, Kp);
  proj_gemm_bf16_kernel<<<gemmGrid, 256, 0, stream>>>(V, Wv, bv, Vp);

  (void)hipFuncSetAttribute(reinterpret_cast<const void*>(attn_fused_kernel),
                            hipFuncAttributeMaxDynamicSharedMemorySize,
                            (int)SMEM_BYTES);
  attn_fused_kernel<<<dim3(SEQ / 16, BATCH), 256, SMEM_BYTES, stream>>>(
      Qp, Kp, Vp, mask, out);
}
